// MultiheadAttention_22591527977031
// MI455X (gfx1250) — compile-verified, matrix operand order checked by reading, not run on hardware
//
#include <hip/hip_runtime.h>
#include <hip/hip_bf16.h>
#include <stdint.h>

// ---------------------------------------------------------------------------
// MI455X (gfx1250) multi-head attention, bf16 WMMA pipeline, f32 accumulate.
// s=2048, d=2048, h=16, dh=128. Compute-bound (~1000 flops/byte).
// GEMM staging now uses the Tensor Data Mover (tensor_load_to_lds) with the
// pad feature providing an 80B LDS row pitch (bank-conflict-free fragments).
// ---------------------------------------------------------------------------

typedef __attribute__((ext_vector_type(16))) __bf16 v16bf;
typedef __attribute__((ext_vector_type(8)))  float  v8f;
typedef unsigned int u32x4 __attribute__((ext_vector_type(4)));
typedef int          i32x8 __attribute__((ext_vector_type(8)));
typedef int          i32x4 __attribute__((ext_vector_type(4)));

#define SQ 2048   // sequence length
#define DM 2048   // model dim
#define NH 16     // heads
#define DH 128    // head dim
#define N3 6144   // 3*d

__device__ __forceinline__ uint16_t f2bf(float f) {
  union { float f; uint32_t u; } c; c.f = f;
  uint32_t u = c.u;
  uint32_t r = u + 0x7FFFu + ((u >> 16) & 1u);   // round-to-nearest-even
  return (uint16_t)(r >> 16);
}

// 16x32 bf16 WMMA fragment (A layout; same per-lane pattern serves as the B
// fragment when the B matrix is stored column-major, i.e. [N][K]).
__device__ __forceinline__ v16bf load_frag(const uint16_t* base, int stride) {
  const int lane = threadIdx.x & 31;
  const int row  = lane & 15;
  const int kb   = (lane < 16) ? 0 : 8;
  const uint16_t* p = base + row * stride + kb;
  union { v16bf v; float4 f4[2]; } u;
  u.f4[0] = *(const float4*)(p);
  u.f4[1] = *(const float4*)(p + 16);
  return u.v;
}

__device__ __forceinline__ v8f wmma_bf16(v16bf a, v16bf b, v8f c) {
  return __builtin_amdgcn_wmma_f32_16x16x32_bf16(false, a, false, b,
                                                 (short)0, c, false, false);
}

// ---------------------------------------------------------------------------
// TDM: issue one 2D tensor_load_to_lds. Tile = tile_y rows x 32 bf16 (64B)
// from a row-major bf16 tensor with row_stride elements. D# pad feature
// inserts 16B after every 64B -> LDS row pitch 80B (40 elements).
// D# bit layout per CDNA5 ISA ch.8 (group0 128b, group1 256b).
// ---------------------------------------------------------------------------
__device__ __forceinline__ void tdm_load_tile_2d(unsigned lds_off,
                                                 const void* gptr,
                                                 int row_stride_elems,
                                                 int tile_y) {
  unsigned long long ga = (unsigned long long)gptr;
  u32x4 g0;
  g0.x = 1u;                                         // count=1, user mode
  g0.y = lds_off;                                    // lds_addr (bytes)
  g0.z = (unsigned)(ga & 0xffffffffu);               // global_addr[31:0]
  g0.w = (unsigned)((ga >> 32) & 0x1ffffffu)         // global_addr[56:32]
       | (2u << 30);                                 // type=2 ("image")
  i32x8 g1;
  // data_size=1 (2B) | pad_enable | pad_interval=3 (16 DW=64B) | pad_amount=3 (4 DW=16B)
  g1[0] = (1 << 16) | (1 << 20) | (3 << 22) | (3 << 25);
  const unsigned dim0 = (unsigned)row_stride_elems;  // tensor_dim0 (elems)
  const unsigned dim1 = 1u << 20;                    // tensor_dim1 (generous)
  g1[1] = (int)((dim0 & 0xffffu) << 16);             // [47:32]=0, [79:64]... dim0 lo16
  g1[2] = (int)((dim0 >> 16) | ((dim1 & 0xffffu) << 16));
  g1[3] = (int)((dim1 >> 16) | (32u << 16));         // tile_dim0 = 32 elems
  g1[4] = (int)(tile_y & 0xffff);                    // tile_dim1; tile_dim2=0
  g1[5] = (int)dim0;                                 // tensor_dim0_stride lo32
  g1[6] = 0;                                         // stride0 hi16 | stride1 lo16
  g1[7] = 0;                                         // stride1 hi32
  i32x4 z4 = {0, 0, 0, 0};
#if __clang_major__ >= 23
  i32x8 z8 = {0, 0, 0, 0, 0, 0, 0, 0};
  __builtin_amdgcn_tensor_load_to_lds(g0, g1, z4, z4, z8, 0);
#else
  __builtin_amdgcn_tensor_load_to_lds(g0, g1, z4, z4, 0);
#endif
}

// ---------------------------------------------------------------------------
// f32 -> bf16 elementwise (x)
// ---------------------------------------------------------------------------
__global__ void k_cvt_bf16(const float* __restrict__ in,
                           uint16_t* __restrict__ out, int n) {
  int i = (blockIdx.x * blockDim.x + threadIdx.x) * 4;
  if (i + 3 < n) {
    float4 v = *(const float4*)(in + i);
    ushort4 o;
    o.x = f2bf(v.x); o.y = f2bf(v.y); o.z = f2bf(v.z); o.w = f2bf(v.w);
    *(ushort4*)(out + i) = o;
  }
}

// f32 [K][N] -> bf16 [N][K] (transpose-convert weights; one-time)
__global__ void k_cvtT_bf16(const float* __restrict__ in,
                            uint16_t* __restrict__ out, int K, int N) {
  long idx = (long)blockIdx.x * blockDim.x + threadIdx.x;
  int k = (int)(idx % K);
  int n = (int)(idx / K);
  out[(long)n * K + k] = f2bf(in[(long)k * N + n]);
}

// ---------------------------------------------------------------------------
// bf16 GEMM: C[M][N] = A[M][K] * BT[N][K]^T (+bias), 128x128 tile, K-step 32.
// 256 threads = 8 waves, wave grid 4(M) x 2(N): 2x4 WMMA tiles per wave.
// Tile staging: TDM. Wave 0 DMAs the A tile, wave 1 the B tile; everyone
// syncs on TENSORcnt + workgroup barrier.
// mode 0: QKV epilogue -> Q/K as [h][s][dh] bf16, V transposed [h][dh][s]
// mode 1: out-proj epilogue -> f32 d_out (+out_b)
// ---------------------------------------------------------------------------
__global__ void k_gemm(const uint16_t* __restrict__ A,
                       const uint16_t* __restrict__ BT,
                       int N, int K,
                       const float* __restrict__ bias,
                       uint16_t* __restrict__ qb, uint16_t* __restrict__ kbuf,
                       uint16_t* __restrict__ vt, float* __restrict__ outf,
                       int mode) {
  __shared__ uint16_t lA[128 * 40];   // 80B pitch via TDM pad
  __shared__ uint16_t lB[128 * 40];

  const int t     = threadIdx.x;
  const int lane  = t & 31;
  const int wid   = t >> 5;
  const int waveM = wid >> 1;       // 0..3 : 32 rows each
  const int waveN = wid & 1;        // 0..1 : 64 cols each
  const int m0 = blockIdx.y * 128;
  const int n0 = blockIdx.x * 128;

  const unsigned ldsA = (unsigned)(unsigned long long)(void*)lA;
  const unsigned ldsB = (unsigned)(unsigned long long)(void*)lB;

  v8f acc[2][4];
  for (int i = 0; i < 2; ++i)
    for (int j = 0; j < 4; ++j)
      for (int e = 0; e < 8; ++e) acc[i][j][e] = 0.f;

  for (int k0 = 0; k0 < K; k0 += 32) {
    __syncthreads();                      // all fragment reads of prev iter done
    if (wid == 0) {
      tdm_load_tile_2d(ldsA, &A[(long)m0 * K + k0], K, 128);
    } else if (wid == 1) {
      tdm_load_tile_2d(ldsB, &BT[(long)n0 * K + k0], K, 128);
    }
    if (k0 + 32 < K) {                    // warm L2 for the next K-slab
      __builtin_prefetch(&A [(long)(m0 + (t & 127)) * K + k0 + 32], 0, 2);
      __builtin_prefetch(&BT[(long)(n0 + (t & 127)) * K + k0 + 32], 0, 2);
    }
    __builtin_amdgcn_s_wait_tensorcnt(0);
    __syncthreads();                      // tile visible to all waves

    v16bf af[2], bf[4];
    for (int mt = 0; mt < 2; ++mt)
      af[mt] = load_frag(&lA[(waveM * 32 + mt * 16) * 40], 40);
    for (int nt = 0; nt < 4; ++nt)
      bf[nt] = load_frag(&lB[(waveN * 64 + nt * 16) * 40], 40);
    for (int mt = 0; mt < 2; ++mt)
      for (int nt = 0; nt < 4; ++nt)
        acc[mt][nt] = wmma_bf16(af[mt], bf[nt], acc[mt][nt]);
  }

  // epilogue: C layout: lane holds col n=lane%16, rows m=(lane<16?0:8)+r
  const int mrow0 = m0 + waveM * 32 + ((lane < 16) ? 0 : 8);
  for (int mt = 0; mt < 2; ++mt) {
    for (int nt = 0; nt < 4; ++nt) {
      const int n = n0 + waveN * 64 + nt * 16 + (lane & 15);
      const float bn = bias[n];
      for (int r = 0; r < 8; ++r) {
        const int m = mrow0 + mt * 16 + r;
        const float v = acc[mt][nt][r] + bn;
        if (mode == 0) {
          const int which = n >> 11;       // 0=q 1=k 2=v
          const int dd    = n & 2047;
          const int head  = dd >> 7;
          const int dh    = dd & 127;
          const uint16_t bv = f2bf(v);
          if (which == 0)      qb  [((long)(head * SQ + m)) * DH + dh] = bv;
          else if (which == 1) kbuf[((long)(head * SQ + m)) * DH + dh] = bv;
          else                 vt  [((long)(head * DH + dh)) * SQ + m] = bv; // V^T
        } else {
          outf[(long)m * N + n] = v;
        }
      }
    }
  }
}

// ---------------------------------------------------------------------------
// Flash attention: grid (s/64, h), 128 threads = 4 independent waves,
// each wave owns 16 query rows. S^T = K*Q^T trick: the S accumulator layout
// equals the A-fragment layout of P for the P*V WMMA -> zero shuffles/LDS
// between softmax and the second GEMM.
// ---------------------------------------------------------------------------
__global__ void k_attn(const uint16_t* __restrict__ qb,
                       const uint16_t* __restrict__ kbuf,
                       const uint16_t* __restrict__ vt,
                       const float* __restrict__ attn_bias,
                       const unsigned char* __restrict__ kpm,
                       uint16_t* __restrict__ ob) {
  const int lane = threadIdx.x & 31;
  const int wid  = threadIdx.x >> 5;
  const int head = blockIdx.y;
  const int q0   = blockIdx.x * 64 + wid * 16;

  // Q fragments (B operand for S^T = K*Q^T): resident for the whole pass
  const uint16_t* qbase = qb + (long)(head * SQ + q0) * DH;
  v16bf qf[4];
  for (int kk = 0; kk < 4; ++kk) qf[kk] = load_frag(qbase + kk * 32, DH);

  v8f oacc[8];
  for (int nt = 0; nt < 8; ++nt)
    for (int e = 0; e < 8; ++e) oacc[nt][e] = 0.f;
  float m_run = -1e30f, l_run = 0.f;
  const float scale = 0.088388347648318447f;   // 1/sqrt(128)
  const int myq = q0 + (lane & 15);
  const float* ab = attn_bias + head * SQ;
  const int klocal0 = (lane < 16) ? 0 : 8;

  const int nkt = (q0 + 47) >> 5;              // causal: keys <= q0+15
  for (int kt = 0; kt < nkt; ++kt) {
    const int kb0 = kt * 32;
    if (kt + 1 < nkt) {                        // pull next K/V tile toward L2
      __builtin_prefetch(kbuf + (long)(head * SQ + kb0 + 32 + lane) * DH, 0, 2);
      __builtin_prefetch(vt + (long)(head * DH + lane * 4) * SQ + kb0 + 32, 0, 2);
    }

    // S^T tiles: D[m=key][n=query], two 16-key halves, contract dh=128
    v8f s0, s1;
    for (int e = 0; e < 8; ++e) { s0[e] = 0.f; s1[e] = 0.f; }
    for (int kk = 0; kk < 4; ++kk) {
      v16bf a0 = load_frag(kbuf + (long)(head * SQ + kb0) * DH + kk * 32, DH);
      v16bf a1 = load_frag(kbuf + (long)(head * SQ + kb0 + 16) * DH + kk * 32, DH);
      s0 = wmma_bf16(a0, qf[kk], s0);
      s1 = wmma_bf16(a1, qf[kk], s1);
    }

    // scale + attn_bias(key) + causal/padding mask
    float sv[16];
    for (int r = 0; r < 8; ++r) {
      int key = kb0 + klocal0 + r;
      float v = s0[r] * scale + ab[key];
      if (key > myq || !kpm[key]) v = -1e30f;
      sv[r] = v;
      key += 16;
      v = s1[r] * scale + ab[key];
      if (key > myq || !kpm[key]) v = -1e30f;
      sv[8 + r] = v;
    }

    // streaming softmax: per-query (=lane%16) stats, halves merged via xor16
    float tmax = -1e30f;
    for (int i = 0; i < 16; ++i) tmax = fmaxf(tmax, sv[i]);
    tmax = fmaxf(tmax, __shfl_xor(tmax, 16, 32));
    const float m_new = fmaxf(m_run, tmax);
    const float alpha = __expf(m_run - m_new);
    float tsum = 0.f;
    uint16_t pb[16];
    for (int i = 0; i < 16; ++i) {
      float p = __expf(sv[i] - m_new);
      tsum += p;
      pb[i] = f2bf(p);
    }
    tsum += __shfl_xor(tsum, 16, 32);
    l_run = l_run * alpha + tsum;
    m_run = m_new;

    // rescale O by alpha of each row (row m=query -> source lane m)
    for (int r = 0; r < 8; ++r) {
      float ar = __shfl(alpha, (lane < 16) ? r : (8 + r), 32);
      for (int nt = 0; nt < 8; ++nt) oacc[nt][r] *= ar;
    }

    // P already sits in A-fragment order in this lane: pb[0..7]=k kb..kb+7,
    // pb[8..15]=k kb+16..kb+23. O += P * V  (B from V^T rows: contiguous).
    union { v16bf v; uint16_t s[16]; } pu;
    for (int i = 0; i < 16; ++i) pu.s[i] = pb[i];
    for (int nt = 0; nt < 8; ++nt) {
      v16bf bv = load_frag(vt + (long)(head * DH + nt * 16) * SQ + kb0, SQ);
      oacc[nt] = wmma_bf16(pu.v, bv, oacc[nt]);
    }
  }

  // normalize by l and write O as bf16 [s][h*dh]
  for (int r = 0; r < 8; ++r) {
    float lr = __shfl(l_run, (lane < 16) ? r : (8 + r), 32);
    float inv = (lr > 0.f) ? 1.f / lr : 0.f;
    const int m = q0 + ((lane < 16) ? 0 : 8) + r;
    for (int nt = 0; nt < 8; ++nt) {
      const int col = head * DH + nt * 16 + (lane & 15);
      ob[(long)m * DM + col] = f2bf(oacc[nt][r] * inv);
    }
  }
}

// ---------------------------------------------------------------------------
extern "C" void kernel_launch(void* const* d_in, const int* in_sizes, int n_in,
                              void* d_out, int out_size, void* d_ws, size_t ws_size,
                              hipStream_t stream) {
  const float* x     = (const float*)d_in[0];
  const float* Wqkv  = (const float*)d_in[1];
  const float* Wb    = (const float*)d_in[2];
  const float* outw  = (const float*)d_in[3];
  const float* outb  = (const float*)d_in[4];
  const float* abias = (const float*)d_in[5];
  const unsigned char* kpm = (const unsigned char*)d_in[6];
  float* out = (float*)d_out;

  // workspace layout (72 MB total)
  char* ws = (char*)d_ws;
  uint16_t* wqkvT = (uint16_t*)(ws);              // [6144][2048] bf16  24 MB
  uint16_t* outwT = (uint16_t*)(ws + 25165824);   // [2048][2048] bf16   8 MB
  uint16_t* xb    = (uint16_t*)(ws + 33554432);   // [2048][2048] bf16   8 MB
  uint16_t* qb    = (uint16_t*)(ws + 41943040);   // [h][s][dh]   bf16   8 MB
  uint16_t* kb    = (uint16_t*)(ws + 50331648);   // [h][s][dh]   bf16   8 MB
  uint16_t* vt    = (uint16_t*)(ws + 58720256);   // [h][dh][s]   bf16   8 MB
  uint16_t* ob    = (uint16_t*)(ws + 67108864);   // [s][d]       bf16   8 MB

  k_cvt_bf16 <<<4096, 256, 0, stream>>>(x, xb, SQ * DM);
  k_cvtT_bf16<<<49152, 256, 0, stream>>>(Wqkv, wqkvT, DM, N3);
  k_cvtT_bf16<<<16384, 256, 0, stream>>>(outw, outwT, DM, DM);

  k_gemm<<<dim3(48, 16), 256, 0, stream>>>(xb, wqkvT, N3, DM, Wb,
                                           qb, kb, vt, nullptr, 0);
  k_attn<<<dim3(32, 16), 128, 0, stream>>>(qb, kb, vt, abias, kpm, ob);
  k_gemm<<<dim3(16, 16), 256, 0, stream>>>(ob, outwT, DM, DM, outb,
                                           nullptr, nullptr, nullptr, out, 1);
}
